// BGNN_41850161332463
// MI455X (gfx1250) — matrix operand order repaired and checked by reading, not compiled, not run on hardware
//
#include <hip/hip_runtime.h>

typedef __attribute__((ext_vector_type(16))) _Float16 v16h;
typedef __attribute__((ext_vector_type(8)))  float    v8f;

#define NNODES 100000
#define NEDGES 3200000
#define NGRAPH 64

// ---------------- utility kernels ----------------

__global__ void k_zero(float* __restrict__ p, int n) {
  int i = blockIdx.x * blockDim.x + threadIdx.x;
  if (i < n) p[i] = 0.0f;
}

__global__ void k_deg(const int* __restrict__ dst, float* __restrict__ deg, int E) {
  int e = blockIdx.x * blockDim.x + threadIdx.x;
  if (e < E) atomicAdd(&deg[dst[e]], 1.0f);
}

__global__ void k_dis(float* __restrict__ deg, int n) {
  int i = blockIdx.x * blockDim.x + threadIdx.x;
  if (i < n) deg[i] = rsqrtf(deg[i] + 1.0f);   // (deg+1)^-0.5, self-loop included
}

__global__ void k_relu(float* __restrict__ p, int n) {
  int i = blockIdx.x * blockDim.x + threadIdx.x;
  if (i < n) p[i] = fmaxf(p[i], 0.0f);
}

// agg[i*ldA + c] = dis_i^2 * h[i*C + c] + b[c]  for c < C, and 0 for C <= c < ldA
// (self-loop + bias init; zero-fills K-padding columns so the next matmul can
//  run with K rounded up to a multiple of 32 and no bounds checks)
__global__ void k_selfinit(const float* __restrict__ h, const float* __restrict__ dis,
                           const float* __restrict__ bias, float* __restrict__ agg,
                           int n, int C, int ldA) {
  int idx = blockIdx.x * blockDim.x + threadIdx.x;
  if (idx >= n * ldA) return;
  int i = idx / ldA, c = idx % ldA;
  float v = 0.0f;
  if (c < C) {
    float d = dis[i];
    v = d * d * h[(size_t)i * C + c] + bias[c];
  }
  agg[idx] = v;
}

// agg[dst*ldA + c] += dis[src]*dis[dst] * h[src*C + c]
// one thread per (edge, 4-channel group); targets are L2-resident (<=12.8MB)
__global__ void k_edge(const int* __restrict__ src, const int* __restrict__ dst,
                       const float* __restrict__ dis, const float* __restrict__ h,
                       int C, float* __restrict__ agg, int ldA, int E, int lgCq) {
  long t = (long)blockIdx.x * blockDim.x + threadIdx.x;
  if (t >= ((long)E << lgCq)) return;
  int e = (int)(t >> lgCq);
  int q = (int)(t & ((1 << lgCq) - 1));
  int s = src[e], d = dst[e];
  float norm = dis[s] * dis[d];
  float4 hv = *(const float4*)(h + (size_t)s * C + q * 4);
  float* ad = agg + (size_t)d * ldA + q * 4;
  atomicAdd(&ad[0], hv.x * norm);
  atomicAdd(&ad[1], hv.y * norm);
  atomicAdd(&ad[2], hv.z * norm);
  atomicAdd(&ad[3], hv.w * norm);
}

// ---------------- WMMA dense transform ----------------
// ISA 7.12.2 layouts for V_WMMA_F32_16X16X32_F16 (wave32):
//  A (16x32 MxK): lane L -> M=L%16 ; K = 8*(L/16) + {0..7} in elems 0..7,
//                                    K = 8*(L/16) + 16 + {0..7} in elems 8..15
//  B (32x16 KxN): lane L -> N=L%16 ; K = 8*(L/16) + {0..7} in elems 0..7, 8*(L/16) + 16 + {0..7} in elems 8..15
//  C/D (16x16 f32): lane L -> N=L%16 ; VGPR r -> M = r + 8*(L/16)

__device__ __forceinline__ int b_kidx(int lane, int i) {
  int j = i >> 1, pos = i & 1;
  return ((lane >= 16) ? 8 : 0) + 2 * j + pos + ((i >= 8) ? 8 : 0);
}

// Bake W[K x C] (row-major) into per-lane B fragments, half precision,
// with zero padding for k >= Klog or n >= C.  frag index:
// ((ks*tilesN + tn)*32 + lane)*16 + i
__global__ void k_prepW(const float* __restrict__ W, int Klog, int C,
                        int Ksteps, int tilesN, _Float16* __restrict__ frag) {
  int t = blockIdx.x * blockDim.x + threadIdx.x;
  int total = Ksteps * tilesN * 32;
  if (t >= total) return;
  int lane = t & 31;
  int tn = (t >> 5) % tilesN;
  int ks = t / (32 * tilesN);
  int n = tn * 16 + (lane & 15);
  _Float16* out = frag + (size_t)t * 16;
#pragma unroll
  for (int i = 0; i < 16; ++i) {
    int k = ks * 32 + b_kidx(lane, i);
    out[i] = (k < Klog && n < C) ? (_Float16)W[(size_t)k * C + n] : (_Float16)0.0f;
  }
}

__device__ __forceinline__ void cvt4(v16h& a, int base, float4 s) {
  a[base + 0] = (_Float16)s.x; a[base + 1] = (_Float16)s.y;
  a[base + 2] = (_Float16)s.z; a[base + 3] = (_Float16)s.w;
}

// H[M x C] = X[M x Kpad] @ W ; Kpad % 32 == 0, M % 16 == 0, branchless hot loop.
__global__ void k_matmul_wmma(const float* __restrict__ X, int ldX,
                              const _Float16* __restrict__ Wfrag,
                              float* __restrict__ H, int ldH,
                              int M, int Kpad, int C, int tilesN) {
  int lane = threadIdx.x & 31;
  int wave = threadIdx.x >> 5;
  int tm = blockIdx.x * (blockDim.x >> 5) + wave;   // 16-row tile per wave
  int tn = blockIdx.y;
  int row0 = tm * 16;
  if (row0 >= M) return;                            // wave-uniform exit

  int mrow = row0 + (lane & 15);
  const float* xrow = X + (size_t)mrow * ldX + ((lane >> 4) << 3); // lane K base
  int nks = Kpad >> 5;

  v8f acc = {};
  for (int ks = 0; ks < nks; ++ks) {
    const float4* px = (const float4*)(xrow + ks * 32);
    float4 s0 = px[0], s1 = px[1];   // lane-relative K 0..7   (b128 x2)
    float4 s2 = px[4], s3 = px[5];   // lane-relative K 16..23 (b128 x2)
    v16h a;
    cvt4(a, 0, s0); cvt4(a, 4, s1); cvt4(a, 8, s2); cvt4(a, 12, s3);
    v16h b = *(const v16h*)(Wfrag + ((size_t)((ks * tilesN + tn) * 32 + lane) << 4));
    acc = __builtin_amdgcn_wmma_f32_16x16x32_f16(false, a, false, b, (short)0, acc,
                                                 false, false);
  }

  int n = tn * 16 + (lane & 15);
  if (n < C) {
    int mbase = row0 + ((lane >> 4) << 3);
#pragma unroll
    for (int r = 0; r < 8; ++r) H[(size_t)(mbase + r) * ldH + n] = acc[r];
  }
}

// ---------------- pooling + MLP head ----------------

__global__ void k_pool(const float* __restrict__ h, const int* __restrict__ batch,
                       float* __restrict__ sums, float* __restrict__ cnt, int n) {
  int i = blockIdx.x * blockDim.x + threadIdx.x;
  if (i >= n) return;
  int g = batch[i];
  for (int c = 0; c < 8; ++c) atomicAdd(&sums[g * 8 + c], h[(size_t)i * 8 + c]);
  atomicAdd(&cnt[g], 1.0f);
}

__global__ void k_mlp(const float* __restrict__ sums0, const float* __restrict__ cnt0,
                      const float* __restrict__ sums1, const float* __restrict__ cnt1,
                      const float* __restrict__ mW1, const float* __restrict__ mb1,
                      const float* __restrict__ mW2, const float* __restrict__ mb2,
                      float* __restrict__ out) {
  int g = threadIdx.x;
  if (g >= NGRAPH) return;
  float x[16];
  float c0 = fmaxf(cnt0[g], 1.0f), c1 = fmaxf(cnt1[g], 1.0f);
  for (int c = 0; c < 8; ++c) {
    x[c]     = sums0[g * 8 + c] / c0;
    x[8 + c] = sums1[g * 8 + c] / c1;
  }
  float hid[8];
  for (int j = 0; j < 8; ++j) {
    float acc = mb1[j];
    for (int k = 0; k < 16; ++k) acc += x[k] * mW1[k * 8 + j];
    hid[j] = fmaxf(acc, 0.0f);
  }
  for (int j = 0; j < 2; ++j) {
    float acc = mb2[j];
    for (int k = 0; k < 8; ++k) acc += hid[k] * mW2[k * 2 + j];
    out[g * 2 + j] = acc;
  }
}

// ---------------- host orchestration ----------------

extern "C" void kernel_launch(void* const* d_in, const int* in_sizes, int n_in,
                              void* d_out, int out_size, void* d_ws, size_t ws_size,
                              hipStream_t stream) {
  (void)in_sizes; (void)n_in; (void)out_size; (void)ws_size;

  const float* x[2]     = {(const float*)d_in[0], (const float*)d_in[1]};
  const int*   ei[2]    = {(const int*)d_in[2],   (const int*)d_in[3]};
  const int*   batch[2] = {(const int*)d_in[4],   (const int*)d_in[5]};
  const float* W[3] = {(const float*)d_in[6], (const float*)d_in[8], (const float*)d_in[10]};
  const float* B[3] = {(const float*)d_in[7], (const float*)d_in[9], (const float*)d_in[11]};
  const float* mW1 = (const float*)d_in[12];
  const float* mb1 = (const float*)d_in[13];
  const float* mW2 = (const float*)d_in[14];
  const float* mb2 = (const float*)d_in[15];

  // workspace: dis | hbuf(N*32) | aggbuf(N*32) | pools | W fragments
  char*  ws     = (char*)d_ws;
  float* dis    = (float*)ws;                            // N floats
  float* hbuf   = (float*)(ws + (size_t)1  * (1 << 20)); // 12.8 MB
  float* aggbuf = (float*)(ws + (size_t)17 * (1 << 20)); // 12.8 MB
  float* psum   = (float*)(ws + (size_t)33 * (1 << 20)); // 2*G*8
  float* pcnt   = psum + 2 * NGRAPH * 8;                 // 2*G
  _Float16* wfrag[3] = {
    (_Float16*)(ws + (size_t)33 * (1 << 20) + (64 << 10)),
    (_Float16*)(ws + (size_t)33 * (1 << 20) + (80 << 10)),
    (_Float16*)(ws + (size_t)33 * (1 << 20) + (96 << 10))};

  // per layer: logical K / padded K / logical C / col tiles / output stride
  const int Klog[3]  = {128, 32, 16};
  const int Kpad[3]  = {128, 32, 32};
  const int Cd[3]    = {32, 16, 8};
  const int tilesN[3]= {2, 1, 1};
  const int ldOut[3] = {32, 32, 8};   // layer-2 output padded to 32 for layer-3 K

  // bake weight fragments (once; shared by both branches)
  for (int l = 0; l < 3; ++l) {
    int ksteps = Kpad[l] >> 5;
    int tot = ksteps * tilesN[l] * 32;
    k_prepW<<<(tot + 63) / 64, 64, 0, stream>>>(W[l], Klog[l], Cd[l], ksteps,
                                                tilesN[l], wfrag[l]);
  }

  int npool = 2 * NGRAPH * 8 + 2 * NGRAPH;
  k_zero<<<(npool + 255) / 256, 256, 0, stream>>>(psum, npool);

  for (int br = 0; br < 2; ++br) {
    const int* src = ei[br];
    const int* dst = ei[br] + NEDGES;   // edge_index row 1

    k_zero<<<(NNODES + 255) / 256, 256, 0, stream>>>(dis, NNODES);
    k_deg<<<(NEDGES + 255) / 256, 256, 0, stream>>>(dst, dis, NEDGES);
    k_dis<<<(NNODES + 255) / 256, 256, 0, stream>>>(dis, NNODES);

    const float* xin = x[br];
    int ldX = 128;
    for (int l = 0; l < 3; ++l) {
      int C = Cd[l], ldA = ldOut[l];

      int tilesM = NNODES / 16;                       // N % 16 == 0
      dim3 grid((tilesM + 7) / 8, tilesN[l]);
      k_matmul_wmma<<<grid, 256, 0, stream>>>(xin, ldX, wfrag[l], hbuf, C,
                                              NNODES, Kpad[l], C, tilesN[l]);

      int nc = NNODES * ldA;
      k_selfinit<<<(nc + 255) / 256, 256, 0, stream>>>(hbuf, dis, B[l], aggbuf,
                                                       NNODES, C, ldA);
      int lgCq = (C == 32) ? 3 : (C == 16) ? 2 : 1;   // C/4 groups, power of 2
      long ethreads = (long)NEDGES << lgCq;
      k_edge<<<(unsigned)((ethreads + 255) / 256), 256, 0, stream>>>(
          src, dst, dis, hbuf, C, aggbuf, ldA, NEDGES, lgCq);
      if (l < 2) k_relu<<<(nc + 255) / 256, 256, 0, stream>>>(aggbuf, nc);

      xin = aggbuf;
      ldX = ldA;
    }

    k_pool<<<(NNODES + 255) / 256, 256, 0, stream>>>(aggbuf, batch[br],
                                                     psum + br * NGRAPH * 8,
                                                     pcnt + br * NGRAPH, NNODES);
  }

  k_mlp<<<1, 64, 0, stream>>>(psum, pcnt, psum + NGRAPH * 8, pcnt + NGRAPH,
                              mW1, mb1, mW2, mb2, (float*)d_out);
}
